// MultiHeadedAttentionBan_84636625534979
// MI455X (gfx1250) — matrix operand order repaired and checked by reading, not compiled, not run on hardware
//
#include <hip/hip_runtime.h>
#include <hip/hip_bf16.h>

// ---------------------------------------------------------------------------
// MultiHeadedAttention "Ban" for MI455X (gfx1250, wave32, WMMA)
// Pipeline: fuse-weights -> 3x GEMM (f16 WMMA, f32 acc) -> flash attention
//           (WMMA scores + online softmax + WMMA P@V) -> output GEMM (f32 out)
// Uses GLOBAL_LOAD_ASYNC_TO_LDS_B128 staging (ASYNCcnt) for the pure-copy
// tiles (flash Kr tile, GEMM weight tile), guarded by __has_builtin.
// ---------------------------------------------------------------------------

typedef __attribute__((ext_vector_type(16))) _Float16 v16h;
typedef __attribute__((ext_vector_type(8)))  _Float16 v8h;
typedef __attribute__((ext_vector_type(8)))  float    v8f;
typedef __attribute__((ext_vector_type(4)))  int      v4i;

#define AS1 __attribute__((address_space(1)))
#define AS3 __attribute__((address_space(3)))

#if defined(__AMDGCN__) && __has_builtin(__builtin_amdgcn_global_load_async_to_lds_b128) && __has_builtin(__builtin_amdgcn_s_wait_asynccnt)
#define USE_ASYNC_COPY 1
#else
#define USE_ASYNC_COPY 0
#endif

// Problem constants
static constexpr int Bc  = 16;
static constexpr int Sc  = 1024;
static constexpr int Dc  = 512;
static constexpr int Hc  = 8;
static constexpr int DKc = 64;
static constexpr int DRc = 192;
static constexpr int DQR = Hc * DRc;   // 1536 fused q/k refined width
static constexpr int Mrows = Bc * Sc;  // 16384

// ---------------------------------------------------------------------------
// Fragment helpers (wave32 WMMA 16x16x32 f16 layouts, per CDNA5 ISA 7.12.2)
//  A (16x32, MxK): lane&15 = row; halves 0..7 -> K = hid*8 + 0..7,
//                  halves 8..15 -> K = 16 + hid*8 + 0..7   (hid = lane>>4)
//  B (32x16, KxN): lane&15 = col; same K mapping as A
//  C/D (16x16):    lane&15 = col; vgpr i -> row = i + hid*8
// ---------------------------------------------------------------------------
__device__ __forceinline__ v16h combine16(v8h lo, v8h hi) {
    v16h r;
#pragma unroll
    for (int i = 0; i < 8; ++i) { r[i] = lo[i]; r[i + 8] = hi[i]; }
    return r;
}

// rowp points at start (K=0) of this lane's A-row; K must be contiguous.
__device__ __forceinline__ v16h load_a_frag(const _Float16* rowp, int lane) {
    const int hid = (lane >> 4) & 1;
    v8h lo = *(const v8h*)(rowp + hid * 8);
    v8h hi = *(const v8h*)(rowp + 16 + hid * 8);
    return combine16(lo, hi);
}

// colp points at start (K=0) of this lane's B-column; K must be contiguous.
__device__ __forceinline__ v16h load_b_frag(const _Float16* colp, int lane) {
    const int hid = (lane >> 4) & 1;
    v8h lo = *(const v8h*)(colp + hid * 8);
    v8h hi = *(const v8h*)(colp + 16 + hid * 8);
    return combine16(lo, hi);
}

__device__ __forceinline__ v8f wmma32(v16h a, v16h b, v8f c) {
    return __builtin_amdgcn_wmma_f32_16x16x32_f16(
        /*neg_a=*/false, a, /*neg_b=*/false, b,
        /*c_mod=*/(short)0, c, /*reuse_a=*/false, /*reuse_b=*/false);
}

// Async 16-byte global -> LDS copy (gfx1250 ASYNCcnt path), guarded.
__device__ __forceinline__ void async_copy_b128(const _Float16* gsrc, _Float16* ldst) {
#if USE_ASYNC_COPY
    __builtin_amdgcn_global_load_async_to_lds_b128(
        (AS1 v4i*)gsrc, (AS3 v4i*)ldst, 0, 0);
#else
    *(v8h*)ldst = *(const v8h*)gsrc;
#endif
}

__device__ __forceinline__ void async_copy_wait() {
#if USE_ASYNC_COPY
    __builtin_amdgcn_s_wait_asynccnt(0);
#endif
}

// ---------------------------------------------------------------------------
// Weight fusion: WT[o][d] = sum_dk W[d][h*64+dk] * Wr[dk][r]   (o = h*RD + r)
// bfused[o]    = br[r] + sum_dk b[h*64+dk] * Wr[dk][r]
// Folds the per-head refine projection into the d_model projection.
// ---------------------------------------------------------------------------
__global__ void fuse_weights_kernel(const float* __restrict__ W,
                                    const float* __restrict__ b,
                                    const float* __restrict__ Wr,
                                    const float* __restrict__ br,
                                    _Float16* __restrict__ WT,
                                    float* __restrict__ bfused,
                                    int RD) {
    const int total = Hc * RD * Dc;
    int idx = blockIdx.x * blockDim.x + threadIdx.x;
    if (idx >= total) return;
    const int d  = idx & (Dc - 1);
    const int o  = idx >> 9;          // Dc == 512 == 2^9
    const int hh = o / RD;
    const int r  = o - hh * RD;
    float acc = 0.f;
#pragma unroll 8
    for (int dk = 0; dk < DKc; ++dk)
        acc += W[(size_t)d * Dc + hh * DKc + dk] * Wr[(size_t)dk * RD + r];
    WT[(size_t)o * Dc + d] = (_Float16)acc;
    if (d == 0) {
        float ba = br[r];
        for (int dk = 0; dk < DKc; ++dk)
            ba += b[hh * DKc + dk] * Wr[(size_t)dk * RD + r];
        bfused[o] = ba;
    }
}

// WoT[n][d] = Wo[d][n] as f16
__global__ void transpose_wo_kernel(const float* __restrict__ Wo,
                                    _Float16* __restrict__ WoT) {
    int idx = blockIdx.x * blockDim.x + threadIdx.x;
    if (idx >= Dc * Dc) return;
    const int n = idx >> 9;
    const int d = idx & (Dc - 1);
    WoT[idx] = (_Float16)Wo[(size_t)d * Dc + n];
}

// ---------------------------------------------------------------------------
// Generic WMMA GEMM: C[M,N] = A[M,K] @ WT[N,K]^T + bias[N]
// 128 threads = 4 waves; WG tile 64x64; wave tile 32x32 (2x2 WMMA frags).
// Weight tile streamed via async global->LDS (pure copy); A tile converted
// f32->f16 through VGPRs, overlapping the async DMA.
// ---------------------------------------------------------------------------
template <bool AF32, bool OUTF32>
__global__ __launch_bounds__(128)
void gemm_wmma_kernel(const void* __restrict__ Ap,
                      const _Float16* __restrict__ WT,
                      const float* __restrict__ bias,
                      void* __restrict__ Cp,
                      int M, int N, int K) {
    __shared__ _Float16 As[64][40];   // 32-wide K slab, +8 pad
    __shared__ _Float16 Bs[64][40];

    const int t    = threadIdx.x;
    const int lane = t & 31;
    const int wave = t >> 5;
    const int r15  = lane & 15;
    const int hid  = lane >> 4;
    const int m0   = blockIdx.y * 64;
    const int n0   = blockIdx.x * 64;
    const int wm   = (wave >> 1) * 32;
    const int wn   = (wave & 1) * 32;

    v8f acc[2][2] = {};

    for (int k0 = 0; k0 < K; k0 += 32) {
        __syncthreads();
        // Weight tile: 64x32 halves = 256 x b128 chunks; 2 per thread, async.
#pragma unroll
        for (int j = 0; j < 2; ++j) {
            int chunk = t * 2 + j;
            int r = chunk >> 2, c = (chunk & 3) * 8;
            async_copy_b128(WT + (size_t)(n0 + r) * K + k0 + c, &Bs[r][c]);
        }
        // A tile: load + (optionally) convert f32->f16, coalesced.
#pragma unroll
        for (int i = 0; i < 16; ++i) {
            int idx = i * 128 + t;
            int r = idx >> 5, c = idx & 31;
            float v;
            if (AF32) v = ((const float*)Ap)[(size_t)(m0 + r) * K + k0 + c];
            else      v = (float)((const _Float16*)Ap)[(size_t)(m0 + r) * K + k0 + c];
            As[r][c] = (_Float16)v;
        }
        async_copy_wait();
        __syncthreads();

        v16h afr[2], bfr[2];
#pragma unroll
        for (int mi = 0; mi < 2; ++mi)
            afr[mi] = load_a_frag(&As[wm + mi * 16 + r15][0], lane);
#pragma unroll
        for (int ni = 0; ni < 2; ++ni)
            bfr[ni] = load_b_frag(&Bs[wn + ni * 16 + r15][0], lane);
#pragma unroll
        for (int mi = 0; mi < 2; ++mi)
#pragma unroll
            for (int ni = 0; ni < 2; ++ni)
                acc[mi][ni] = wmma32(afr[mi], bfr[ni], acc[mi][ni]);
    }

    // Epilogue: bias + store (f16 activations or f32 final output)
#pragma unroll
    for (int mi = 0; mi < 2; ++mi)
#pragma unroll
        for (int ni = 0; ni < 2; ++ni)
#pragma unroll
            for (int i = 0; i < 8; ++i) {
                int row = m0 + wm + mi * 16 + i + hid * 8;
                int col = n0 + wn + ni * 16 + r15;
                float v = acc[mi][ni][i] + bias[col];
                if (OUTF32) ((float*)Cp)[(size_t)row * N + col] = v;
                else        ((_Float16*)Cp)[(size_t)row * N + col] = (_Float16)v;
            }
}

// ---------------------------------------------------------------------------
// Flash attention: grid (S/64, B*H), 128 threads (4 waves).
// Wave w owns 16 query rows. Qr resident as 6 A-fragments (K = 192).
// Per 64-key block: 24 WMMA (scores) + online softmax + 8 WMMA (P@V).
// Kr tile streamed via async global->LDS, overlapped with the V-transpose.
// Writes merged-head f16 x[B,S,512] for the output GEMM.
// ---------------------------------------------------------------------------
__global__ __launch_bounds__(128)
void flash_attn_kernel(const _Float16* __restrict__ qr,
                       const _Float16* __restrict__ kr,
                       const _Float16* __restrict__ vr,
                       const int* __restrict__ mask,
                       _Float16* __restrict__ xout) {
    __shared__ _Float16 Ks[64][DRc + 8];  // [key][feat], natural B-layout for QK^T
    __shared__ _Float16 Vt[64][72];       // [feat][key], B-layout for P@V
    __shared__ float    Sld[4][16][72];   // per-wave f32 score rows
    __shared__ _Float16 Pld[4][16][72];   // per-wave f16 probability rows
    __shared__ float    stm[4][16], stl[4][16], sta[4][16];

    const int t    = threadIdx.x;
    const int lane = t & 31;
    const int wave = t >> 5;
    const int r15  = lane & 15;
    const int hid  = lane >> 4;
    const int bh   = blockIdx.y;
    const int b    = bh >> 3;      // H == 8
    const int h    = bh & 7;
    const int q0   = blockIdx.x * 64;
    const int qw   = q0 + wave * 16;

    if (lane < 16) { stm[wave][lane] = -3.0e38f; stl[wave][lane] = 0.f; sta[wave][lane] = 1.f; }

    // Resident Q fragments: K = 192 -> 6 chunks of 32
    const _Float16* qbase = qr + ((size_t)(b * Sc) + qw + r15) * DQR + h * DRc;
    v16h qf[6];
#pragma unroll
    for (int c = 0; c < 6; ++c) qf[c] = load_a_frag(qbase + c * 32, lane);

    v8f of[4] = {};

    for (int kb = 0; kb < Sc; kb += 64) {
        __syncthreads();
        // Kr block: 64 keys x 192 feats; thread -> (key, 96-feat half).
        // Pure copy: route through async global->LDS (12 x b128 per thread).
        {
            const int key = t >> 1, f0 = (t & 1) * 96;
            const _Float16* src = kr + ((size_t)(b * Sc) + kb + key) * DQR + h * DRc + f0;
            _Float16* dst = &Ks[key][f0];
#pragma unroll
            for (int i = 0; i < 12; ++i)
                async_copy_b128(src + i * 8, dst + i * 8);
        }
        // V block transposed into Vt[feat][key] (conversion of layout -> VGPR path)
        {
            const int key = t & 63, fb = (t >> 6) * 32;
            const _Float16* src = vr + ((size_t)(b * Sc) + kb + key) * Dc + h * DKc + fb;
#pragma unroll
            for (int i = 0; i < 32; ++i) Vt[fb + i][key] = src[i];
        }
        async_copy_wait();
        __syncthreads();

        // Scores: 4 key sub-tiles x 6 K-chunks
        v8f sf[4];
#pragma unroll
        for (int n = 0; n < 4; ++n) {
            v8f s = {};
#pragma unroll
            for (int c = 0; c < 6; ++c) {
                v16h bf = load_b_frag(&Ks[n * 16 + r15][c * 32], lane);
                s = wmma32(qf[c], bf, s);
            }
            sf[n] = s;
        }
        // scale 1/sqrt(64), mask, stage f32 rows in LDS
#pragma unroll
        for (int n = 0; n < 4; ++n)
#pragma unroll
            for (int i = 0; i < 8; ++i) {
                const int rr = i + hid * 8;
                const int kc = kb + n * 16 + r15;
                float v = sf[n][i] * 0.125f;
                if (mask[((size_t)b * Sc + (qw + rr)) * Sc + kc] == 0) v = -1e9f;
                Sld[wave][rr][n * 16 + r15] = v;
            }
        __syncthreads();

        // Online softmax update (one lane per query row)
        if (lane < 16) {
            const int rr = lane;
            float mold = stm[wave][rr];
            float mx = mold;
            for (int j = 0; j < 64; ++j) mx = fmaxf(mx, Sld[wave][rr][j]);
            float alpha = __expf(mold - mx);
            float sum = 0.f;
            for (int j = 0; j < 64; ++j) {
                float p = __expf(Sld[wave][rr][j] - mx);
                Pld[wave][rr][j] = (_Float16)p;
                sum += p;
            }
            stm[wave][rr] = mx;
            stl[wave][rr] = stl[wave][rr] * alpha + sum;
            sta[wave][rr] = alpha;
        }
        __syncthreads();

        // Rescale running O, then accumulate P@V
        float al[8];
#pragma unroll
        for (int i = 0; i < 8; ++i) al[i] = sta[wave][i + hid * 8];
#pragma unroll
        for (int n = 0; n < 4; ++n)
#pragma unroll
            for (int i = 0; i < 8; ++i) of[n][i] *= al[i];

#pragma unroll
        for (int kc = 0; kc < 2; ++kc) {
            v16h pf = load_a_frag(&Pld[wave][r15][kc * 32], lane);
#pragma unroll
            for (int nf = 0; nf < 4; ++nf) {
                v16h vf = load_b_frag(&Vt[nf * 16 + r15][kc * 32], lane);
                of[nf] = wmma32(pf, vf, of[nf]);
            }
        }
    }

    __syncthreads();
    float linv[8];
#pragma unroll
    for (int i = 0; i < 8; ++i) linv[i] = 1.f / stl[wave][i + hid * 8];
#pragma unroll
    for (int nf = 0; nf < 4; ++nf)
#pragma unroll
        for (int i = 0; i < 8; ++i) {
            const int row = qw + i + hid * 8;
            const int col = h * DKc + nf * 16 + r15;
            xout[((size_t)b * Sc + row) * Dc + col] = (_Float16)(of[nf][i] * linv[i]);
        }
}

// ---------------------------------------------------------------------------
// Host launch
// ---------------------------------------------------------------------------
extern "C" void kernel_launch(void* const* d_in, const int* in_sizes, int n_in,
                              void* d_out, int out_size, void* d_ws, size_t ws_size,
                              hipStream_t stream) {
    (void)in_sizes; (void)n_in; (void)out_size; (void)ws_size;
    const float* query = (const float*)d_in[0];
    const float* key   = (const float*)d_in[1];
    const float* value = (const float*)d_in[2];
    const int*   mask  = (const int*)d_in[3];
    const float* Wq  = (const float*)d_in[4];
    const float* bq  = (const float*)d_in[5];
    const float* Wk  = (const float*)d_in[6];
    const float* bk  = (const float*)d_in[7];
    const float* Wv  = (const float*)d_in[8];
    const float* bv  = (const float*)d_in[9];
    const float* Wqr = (const float*)d_in[10];
    const float* bqr = (const float*)d_in[11];
    const float* Wkr = (const float*)d_in[12];
    const float* bkr = (const float*)d_in[13];
    const float* Wvr = (const float*)d_in[14];
    const float* bvr = (const float*)d_in[15];
    const float* Wo  = (const float*)d_in[16];
    const float* bo  = (const float*)d_in[17];

    char* ws = (char*)d_ws;
    size_t off = 0;
    auto take = [&](size_t bytes) -> char* {
        char* p = ws + off;
        off = (off + bytes + 255) & ~(size_t)255;
        return p;
    };
    _Float16* wqT = (_Float16*)take((size_t)DQR * Dc * 2);
    _Float16* wkT = (_Float16*)take((size_t)DQR * Dc * 2);
    _Float16* wvT = (_Float16*)take((size_t)Dc * Dc * 2);
    _Float16* woT = (_Float16*)take((size_t)Dc * Dc * 2);
    float* bqf = (float*)take((size_t)DQR * 4);
    float* bkf = (float*)take((size_t)DQR * 4);
    float* bvf = (float*)take((size_t)Dc * 4);
    _Float16* qrb = (_Float16*)take((size_t)Mrows * DQR * 2);
    _Float16* krb = (_Float16*)take((size_t)Mrows * DQR * 2);
    _Float16* vrb = (_Float16*)take((size_t)Mrows * Dc * 2);
    _Float16* xb  = (_Float16*)take((size_t)Mrows * Dc * 2);

    // 1) Fold refine projections into the d_model projections (f16 weights)
    {
        int tq = Hc * DRc * Dc;
        fuse_weights_kernel<<<(tq + 255) / 256, 256, 0, stream>>>(Wq, bq, Wqr, bqr, wqT, bqf, DRc);
        fuse_weights_kernel<<<(tq + 255) / 256, 256, 0, stream>>>(Wk, bk, Wkr, bkr, wkT, bkf, DRc);
        int tv = Hc * DKc * Dc;
        fuse_weights_kernel<<<(tv + 255) / 256, 256, 0, stream>>>(Wv, bv, Wvr, bvr, wvT, bvf, DKc);
        transpose_wo_kernel<<<(Dc * Dc + 255) / 256, 256, 0, stream>>>(Wo, woT);
    }

    // 2) Fused projections: qr/kr [M,1536], vr [M,512], all f16 WMMA
    {
        dim3 gqk(DQR / 64, Mrows / 64);
        gemm_wmma_kernel<true, false><<<gqk, 128, 0, stream>>>(query, wqT, bqf, qrb, Mrows, DQR, Dc);
        gemm_wmma_kernel<true, false><<<gqk, 128, 0, stream>>>(key,   wkT, bkf, krb, Mrows, DQR, Dc);
        dim3 gv(Dc / 64, Mrows / 64);
        gemm_wmma_kernel<true, false><<<gv, 128, 0, stream>>>(value, wvT, bvf, vrb, Mrows, Dc, Dc);
    }

    // 3) Flash attention (never materializes [B,H,S,S])
    {
        dim3 ga(Sc / 64, Bc * Hc);
        flash_attn_kernel<<<ga, 128, 0, stream>>>(qrb, krb, vrb, mask, xb);
    }

    // 4) Output projection: out f32 = x @ Wo + bo
    {
        dim3 go(Dc / 64, Mrows / 64);
        gemm_wmma_kernel<false, true><<<go, 128, 0, stream>>>(xb, woT, bo, (float*)d_out, Mrows, Dc, Dc);
    }
}
